// NeuralMemory_87299505258997
// MI455X (gfx1250) — compile-verified, matrix operand order checked by reading, not run on hardware
//
#include <hip/hip_runtime.h>
#include <hip/hip_bf16.h>

// ---------------------------------------------------------------------------
// Neural-memory forward for MI455X (gfx1250), wave32 + WMMA f32<-f16 + TDM.
// b=4 t=8192 dim=512 heads=8 d=c=64 n=128 bh=32
// ---------------------------------------------------------------------------

typedef __attribute__((ext_vector_type(16))) _Float16 v16h;
typedef __attribute__((ext_vector_type(8)))  float    v8f;
typedef unsigned int u32x4 __attribute__((ext_vector_type(4)));
typedef int          i32x4 __attribute__((ext_vector_type(4)));
typedef int          i32x8 __attribute__((ext_vector_type(8)));

#define EPSF 1.1920929e-07f
#define MAX_LR 0.01f

// ---------------------------------------------------------------------------
// Tensor Data Mover: 2D tile load Global -> LDS (data_size = 4B).
// Descriptor packed per CDNA5 ISA 8.3 (group0) / 8.4 (group1).
// Issued by one wave; completion via s_wait_tensorcnt.
// ---------------------------------------------------------------------------
__device__ __forceinline__ void tdm_load_2d(void* lds_dst, const float* gsrc,
                                            unsigned tensor_d0, unsigned tensor_d1,
                                            unsigned tile_d0, unsigned tile_d1,
                                            unsigned long long d0_stride /*elements*/) {
    unsigned long long ga = (unsigned long long)(size_t)gsrc;
    unsigned lds = (unsigned)(size_t)lds_dst;     // LDS aperture: addr[31:0] = LDS byte offset
    u32x4 g0;
    g0[0] = 1u;                                   // count=1, user descriptor, no gather
    g0[1] = lds;                                  // lds_addr (bytes)
    g0[2] = (unsigned)(ga & 0xFFFFFFFFu);         // global_addr[31:0]
    g0[3] = (unsigned)((ga >> 32) & 0x01FFFFFFu)  // global_addr[56:32]
          | 0x80000000u;                          // type=2 ("image") in bits[127:126]
    i32x8 g1;
    g1[0] = (int)(2u << 16);                                              // data_size=4B
    g1[1] = (int)((tensor_d0 & 0xFFFFu) << 16);                           // tensor_dim0[15:0]
    g1[2] = (int)((tensor_d0 >> 16) | ((tensor_d1 & 0xFFFFu) << 16));     // td0[31:16]|td1[15:0]
    g1[3] = (int)((tensor_d1 >> 16) | ((tile_d0 & 0xFFFFu) << 16));       // td1[31:16]|tile_d0
    g1[4] = (int)(tile_d1 & 0xFFFFu);                                     // tile_d1, tile_d2=0
    g1[5] = (int)(unsigned)(d0_stride & 0xFFFFFFFFu);                     // dim0_stride[31:0]
    g1[6] = (int)(unsigned)((d0_stride >> 32) & 0xFFFFu);                 // dim0_stride[47:32]
    g1[7] = 0;                                                            // dim1_stride unused (2D tile)
    i32x4 z4 = {0, 0, 0, 0};
#if defined(__clang_major__) && __clang_major__ >= 23
    i32x8 z8 = {0, 0, 0, 0, 0, 0, 0, 0};
    __builtin_amdgcn_tensor_load_to_lds(g0, g1, z4, z4, z8, 0);
#else
    __builtin_amdgcn_tensor_load_to_lds(g0, g1, z4, z4, 0);
#endif
}

// K index inside a 16x32 WMMA fragment, per CDNA5 ISA 16-bit A layout:
// VGPR j holds K = {2(j%4), 2(j%4)+1} + (j>=4 ? 16 : 0) + (lane>=16 ? 8 : 0)
__device__ __forceinline__ int wmma_k_of(int e, int lane) {
    int j = e >> 1, hi = e & 1;
    return ((j & 4) << 2) | ((j & 3) << 1) | hi | ((lane & 16) ? 8 : 0);
}

// A fragment (16x32, M x K).  Logical A[M][K]:
//   !trans: src[(bm+m)*lda + (bk+k)]   trans: src[(bk+k)*lda + (bm+m)]
__device__ __forceinline__ v16h fragA_f32(const float* src, int lda, int bm, int bk, bool trans) {
    int lane = threadIdx.x & 31;
    int m = bm + (lane & 15);
    v16h a;
#pragma unroll
    for (int e = 0; e < 16; ++e) {
        int k = bk + wmma_k_of(e, lane);
        float f = trans ? src[k * lda + m] : src[m * lda + k];
        a[e] = (_Float16)f;
    }
    return a;
}

// B fragment (32x16, K x N).  Logical B[K][N]:
//   !trans: src[(bk+k)*ldb + (bn+n)]   trans: src[(bn+n)*ldb + (bk+k)]
__device__ __forceinline__ v16h fragB_f32(const float* src, int ldb, int bk, int bn, bool trans) {
    int lane = threadIdx.x & 31;
    int n = bn + (lane & 15);
    v16h b;
#pragma unroll
    for (int e = 0; e < 16; ++e) {
        int k = bk + wmma_k_of(e, lane);
        float f = trans ? src[n * ldb + k] : src[k * ldb + n];
        b[e] = (_Float16)f;
    }
    return b;
}

// D(64x64) = dsign * op(A)*op(B)  (64x64x64, f32 accum).  256 threads
// (8 waves, 2 16x16 tiles each).  A/B LDS or L2-resident global f32;
// D may be LDS or global (row stride 64).
__device__ __forceinline__ void mm64g(const float* A, int lda, bool tA,
                                      const float* B, int ldb, bool tB,
                                      float* D, float dsign) {
    int lane = threadIdx.x & 31;
    int wave = threadIdx.x >> 5;
#pragma unroll
    for (int tt = 0; tt < 2; ++tt) {
        int ti = wave + (tt << 3);
        int bm = (ti >> 2) << 4, bn = (ti & 3) << 4;
        v8f acc = {0.f, 0.f, 0.f, 0.f, 0.f, 0.f, 0.f, 0.f};
#pragma unroll
        for (int kb = 0; kb < 64; kb += 32) {
            v16h a = fragA_f32(A, lda, bm, kb, tA);
            v16h b = fragB_f32(B, ldb, kb, bn, tB);
            acc = __builtin_amdgcn_wmma_f32_16x16x32_f16(false, a, false, b,
                                                         (short)0, acc, false, false);
        }
        int mrow = bm + ((lane >> 4) << 3);
        int ncol = bn + (lane & 15);
#pragma unroll
        for (int j = 0; j < 8; ++j) D[(mrow + j) * 64 + ncol] = dsign * acc[j];
    }
}

__device__ __forceinline__ float sigmoidf_(float x) { return 1.f / (1.f + __expf(-x)); }

// ---------------------------------------------------------------------------
// Stage kernels
// ---------------------------------------------------------------------------

// rmsnorm over dim=512; one block per row
__global__ __launch_bounds__(128) void k_rmsnorm(const float* __restrict__ x, float* __restrict__ y) {
    __shared__ float red[128];
    long long row = blockIdx.x;
    const float* xr = x + row * 512;
    float ss = 0.f;
    for (int c = threadIdx.x; c < 512; c += 128) { float v = xr[c]; ss += v * v; }
    red[threadIdx.x] = ss;
    __syncthreads();
    for (int s = 64; s > 0; s >>= 1) {
        if (threadIdx.x < s) red[threadIdx.x] += red[threadIdx.x + s];
        __syncthreads();
    }
    float scale = rsqrtf(red[0] * (1.f / 512.f) + EPSF);
    for (int c = threadIdx.x; c < 512; c += 128) y[row * 512 + c] = xr[c] * scale;
}

// Generic tiled WMMA GEMM: C(MxN) = A(MxK) * B(KxN), f32 in/out, f16 compute.
// 64x64 C tile per 256-thread block.  LDS tiles are staged in WMMA *fragment
// order* ([tile-group][lane][16 halves], 32B aligned) so every fragment load
// is exactly two ds_load_b128 — no scalar u16 scatter on the WMMA issue path.
// shift_mode==1 applies the causal 63-row shift+crop while storing.
__global__ __launch_bounds__(256) void k_gemm(const float* __restrict__ A, const float* __restrict__ B,
                                              float* __restrict__ C, int N, int K, int shift_mode) {
    __shared__ __align__(32) _Float16 sA[2048];   // 4 M-groups x 32 lanes x 16
    __shared__ __align__(32) _Float16 sB[2048];   // 4 N-groups x 32 lanes x 16
    int tid = threadIdx.x;
    int lane = tid & 31, wave = tid >> 5;
    long long bm0 = (long long)blockIdx.y * 64;
    long long bn0 = (long long)blockIdx.x * 64;
    v8f acc[2];
#pragma unroll
    for (int tt = 0; tt < 2; ++tt) acc[tt] = (v8f){0.f, 0.f, 0.f, 0.f, 0.f, 0.f, 0.f, 0.f};

    for (int k0 = 0; k0 < K; k0 += 32) {
        __syncthreads();
#pragma unroll
        for (int u = 0; u < 8; ++u) {             // stage fragment-ordered tiles
            int idx = tid + (u << 8);             // 0..2047
            int e = idx & 15, l = (idx >> 4) & 31, g = idx >> 9;
            int kk = k0 + wmma_k_of(e, l);
            int mn = (g << 4) + (l & 15);
            sA[idx] = (_Float16)A[(bm0 + mn) * (long long)K + kk];
            sB[idx] = (_Float16)B[(long long)kk * N + bn0 + mn];
        }
        __syncthreads();
#pragma unroll
        for (int tt = 0; tt < 2; ++tt) {
            int ti = wave + (tt << 3);
            int gm = ti >> 2, gn = ti & 3;
            v16h a = *(const v16h*)&sA[(((gm << 5) + lane) << 4)];
            v16h b = *(const v16h*)&sB[(((gn << 5) + lane) << 4)];
            acc[tt] = __builtin_amdgcn_wmma_f32_16x16x32_f16(false, a, false, b,
                                                             (short)0, acc[tt], false, false);
        }
    }
#pragma unroll
    for (int tt = 0; tt < 2; ++tt) {
        int ti = wave + (tt << 3);
        int bm = (ti >> 2) << 4, bn = (ti & 3) << 4;
        int mrow = bm + ((lane >> 4) << 3);
        long long ncol = bn0 + bn + (lane & 15);
#pragma unroll
        for (int j = 0; j < 8; ++j) {
            long long r = bm0 + mrow + j;
            if (shift_mode == 0) {
                C[r * N + ncol] = acc[tt][j];
            } else {
                long long bb = r >> 13, jj = r & 8191;       // row within batch
                if (jj <= 8128)                               // j+63 < 8192
                    C[((bb << 13) + jj + 63) * N + ncol] = acc[tt][j];
            }
        }
    }
}

// lr = sigmoid(s@Wstep)*MAX_LR, gate = sigmoid(s@Wgate); one thread per (row,head)
__global__ void k_lr_gate(const float* __restrict__ s, const float* __restrict__ Wstep,
                          const float* __restrict__ Wgate, float* __restrict__ lr,
                          float* __restrict__ gate) {
    long long idx = (long long)blockIdx.x * blockDim.x + threadIdx.x;
    if (idx >= (long long)32768 * 8) return;
    long long row = idx >> 3; int h = idx & 7;
    const float* sr = s + row * 512;
    float a1 = 0.f, a2 = 0.f;
    for (int d = 0; d < 512; ++d) {
        float v = sr[d];
        a1 += v * Wstep[d * 8 + h];
        a2 += v * Wgate[d * 8 + h];
    }
    lr[idx] = sigmoidf_(a1) * MAX_LR;
    gate[idx] = sigmoidf_(a2);
}

// chunk means of s: s_chunk[b][n][d]
__global__ void k_chunk_mean(const float* __restrict__ s, float* __restrict__ s_chunk) {
    long long idx = (long long)blockIdx.x * blockDim.x + threadIdx.x;
    if (idx >= 262144) return;                    // 4*128*512
    int dc = idx & 511;
    long long bn = idx >> 9;                      // b*128+ni
    long long b = bn >> 7, ni = bn & 127;
    float acc = 0.f;
    for (int i = 0; i < 64; ++i)
        acc += s[(b * 8192 + ni * 64 + i) * 512 + dc];
    s_chunk[idx] = acc * (1.f / 64.f);
}

// per-chunk momentum gate and (1-decay)
__global__ void k_gates(const float* __restrict__ s_chunk, const float* __restrict__ Wmom,
                        const float* __restrict__ Wdecay, float* __restrict__ momg,
                        float* __restrict__ omd) {
    int idx = blockIdx.x * blockDim.x + threadIdx.x;
    if (idx >= 4096) return;                      // bh * n
    int bh = idx >> 7, ni = idx & 127;
    int b = bh >> 3, h = bh & 7;
    const float* sc = s_chunk + (((long long)b * 128 + ni) << 9);
    float am = 0.f, ad = 0.f;
    for (int d = 0; d < 512; ++d) {
        float v = sc[d];
        am += v * Wmom[d * 8 + h];
        ad += v * Wdecay[d * 8 + h];
    }
    momg[idx] = sigmoidf_(am);
    omd[idx] = 1.f - sigmoidf_(ad);
}

// Per-chunk tiny-MLP backward: writes surprises (-g1, -g2) directly from
// WMMA accumulators to HBM.  Kc tile staged via the Tensor Data Mover.
// X = Kc@W1, H=silu(X), Y=H@W2, dY=(2/d)*lr*(Y-V),
// G2=H^T@dY, dH=dY@W2^T, dX=dH*silu'(X), G1=Kc^T@dX.
__global__ __launch_bounds__(256) void k_chunk_grad(const float* __restrict__ kv,
        const float* __restrict__ lr_full, const float* __restrict__ w1,
        const float* __restrict__ w2, float* __restrict__ g1buf, float* __restrict__ g2buf) {
    __shared__ float sK[4096], sX[4096], sH[4096], sD[4096];   // 64 KB
    int chunk = blockIdx.x;
    int bh = chunk >> 7, ni = chunk & 127;
    int b = bh >> 3, h = bh & 7;
    int tid = threadIdx.x;
    long long rowbase = (long long)b * 8192 + (long long)ni * 64;
    const float* Kc = kv + rowbase * 1024 + h * 64;   // row stride 1024
    const float* Vc = Kc + 512;
    long long ub = (long long)chunk << 12;

    if (tid < 32) {                                            // wave 0: TDM 64x64 tile
        tdm_load_2d(sK, Kc, 64, 64, 64, 64, 1024ull);
        __builtin_amdgcn_s_wait_tensorcnt(0);
    }
    __syncthreads();

    mm64g(sK, 64, false, w1, 64, false, sX, 1.f);              // X
    __syncthreads();
    for (int e = tid; e < 4096; e += 256) { float x = sX[e]; sH[e] = x * sigmoidf_(x); }
    __syncthreads();
    mm64g(sH, 64, false, w2, 64, false, sD, 1.f);              // Y
    __syncthreads();
    for (int e = tid; e < 4096; e += 256) {
        int i = e >> 6, dc = e & 63;
        float vc = Vc[i * 1024 + dc];
        float lr = lr_full[(rowbase + i) * 8 + h];
        sD[e] = 0.03125f * lr * (sD[e] - vc);                  // dY, 2/d = 1/32
    }
    __syncthreads();
    mm64g(sH, 64, true, sD, 64, false, g2buf + ub, -1.f);      // -G2 -> HBM
    __syncthreads();
    mm64g(sD, 64, false, w2, 64, true, sH, 1.f);               // dH (H dead now)
    __syncthreads();
    for (int e = tid; e < 4096; e += 256) {
        float x = sX[e];
        float sg = sigmoidf_(x);
        sH[e] *= sg * (1.f + x * (1.f - sg));                  // dX
    }
    __syncthreads();
    mm64g(sK, 64, true, sH, 64, false, g1buf + ub, -1.f);      // -G1 -> HBM
}

// fused double scan: m_t = a_t m_{t-1} + g_t ; u_t = b_t u_{t-1} + m_t (in place)
// Next-chunk prefetch hides latency of the only serial dependency chain.
__global__ void k_scan(float* __restrict__ g, const float* __restrict__ momg,
                       const float* __restrict__ omd) {
    long long idx = (long long)blockIdx.x * blockDim.x + threadIdx.x;
    if (idx >= (long long)32 * 4096) return;
    long long bh = idx >> 12; int e = idx & 4095;
    float m = 0.f, u = 0.f;
    for (int ni = 0; ni < 128; ++ni) {
        long long gi = (((bh << 7) + ni) << 12) + e;
        if (ni + 1 < 128) __builtin_prefetch(&g[gi + 4096], 0, 1);
        float a = momg[(bh << 7) + ni];
        float bb = omd[(bh << 7) + ni];
        m = a * m + g[gi];
        u = bb * u + m;
        g[gi] = u;
    }
}

// Retrieve: per chunk, Y = silu(Q@(W1+U1))@(W2+U2), rmsnorm rows, *(gamma+1)*gate.
// Update blocks and the shifted Q tile staged via TDM; the causal pad is
// implemented by TDM OOB zero-fill (tensor_dim1 = remaining valid rows).
__global__ __launch_bounds__(256) void k_retrieve(const float* __restrict__ q,
        const float* __restrict__ w1, const float* __restrict__ w2,
        const float* __restrict__ g1, const float* __restrict__ g2,
        const float* __restrict__ gate, const float* __restrict__ gamma,
        float* __restrict__ out_pre) {
    __shared__ float sW1p[4096], sW2p[4096], sQ[4096], sX[4096];   // 64 KB
    int chunk = blockIdx.x;
    int bh = chunk >> 7, ni = chunk & 127;
    int b = bh >> 3, h = bh & 7;
    int tid = threadIdx.x;
    long long ub = (long long)chunk << 12;
    long long srow0 = (long long)ni * 64 + 63;                 // first shifted s-row

    if (tid < 32) {                                            // wave 0: three TDM tiles
        tdm_load_2d(sW1p, g1 + ub, 4096, 1, 4096, 1, 4096ull);
        tdm_load_2d(sW2p, g2 + ub, 4096, 1, 4096, 1, 4096ull);
        unsigned rows_valid = (unsigned)(8129 - ni * 64);      // rows past t zero-filled
        tdm_load_2d(sQ, q + ((long long)b * 8192 + srow0) * 512 + h * 64,
                    64, rows_valid, 64, 64, 512ull);
        __builtin_amdgcn_s_wait_tensorcnt(0);
    }
    __syncthreads();
    for (int e = tid; e < 4096; e += 256) {                    // fast weights
        sW1p[e] += w1[e];
        sW2p[e] += w2[e];
    }
    __syncthreads();
    mm64g(sQ, 64, false, sW1p, 64, false, sX, 1.f);
    __syncthreads();
    for (int e = tid; e < 4096; e += 256) { float x = sX[e]; sX[e] = x * sigmoidf_(x); }
    __syncthreads();
    mm64g(sX, 64, false, sW2p, 64, false, sQ, 1.f);            // Y -> sQ
    __syncthreads();
    if (tid < 64) {                                            // row rms scales -> sX[0..63]
        float ss = 0.f;
        for (int dc = 0; dc < 64; ++dc) { float v = sQ[tid * 64 + dc]; ss += v * v; }
        sX[tid] = rsqrtf(ss * (1.f / 64.f) + EPSF);
    }
    __syncthreads();
    for (int e = tid; e < 4096; e += 256) {
        int i = e >> 6, dc = e & 63;
        long long j = (long long)ni * 64 + i;
        long long srow = j + 63;
        float gt = (srow < 8192) ? gate[((long long)b * 8192 + srow) * 8 + h] : 0.f;
        float val = sQ[e] * sX[i] * (gamma[h * 64 + dc] + 1.f) * gt;
        out_pre[((long long)b * 8192 + j) * 512 + h * 64 + dc] = val;
    }
}

// zero first c-1 rows of each batch in final output
__global__ void k_zero_head(float* __restrict__ out) {
    long long idx = (long long)blockIdx.x * blockDim.x + threadIdx.x;
    if (idx >= (long long)4 * 63 * 512) return;
    long long b = idx / (63 * 512);
    long long r = idx % (63 * 512);
    out[b * 8192 * 512 + r] = 0.f;
}

// ---------------------------------------------------------------------------
// Workspace layout (float offsets)
// ---------------------------------------------------------------------------
#define OFF_S    0LL            // 16777216  rmsnorm(seq)
#define OFF_KV   16777216LL     // 33554432  s @ Wkv
#define OFF_Q    50331648LL     // 16777216  s @ Wq
#define OFF_G1   67108864LL     // 16777216  surprises->updates (w1)
#define OFF_G2   83886080LL     // 16777216  surprises->updates (w2)
#define OFF_OPRE 100663296LL    // 16777216  merged-head retrieve values
#define OFF_LR   117440512LL    // 262144
#define OFF_GATE 117702656LL    // 262144
#define OFF_SCH  117964800LL    // 262144    chunk means
#define OFF_MOMG 118226944LL    // 4096
#define OFF_OMD  118231040LL    // 4096

extern "C" void kernel_launch(void* const* d_in, const int* in_sizes, int n_in,
                              void* d_out, int out_size, void* d_ws, size_t ws_size,
                              hipStream_t stream) {
    const float* seq   = (const float*)d_in[0];
    const float* w1    = (const float*)d_in[1];
    const float* w2    = (const float*)d_in[2];
    const float* Wq    = (const float*)d_in[3];
    const float* Wkv   = (const float*)d_in[4];
    const float* Wstep = (const float*)d_in[5];
    const float* Wmom  = (const float*)d_in[6];
    const float* Wdec  = (const float*)d_in[7];
    const float* Wgate = (const float*)d_in[8];
    const float* Wcomb = (const float*)d_in[9];
    const float* gamma = (const float*)d_in[10];
    float* out = (float*)d_out;
    float* ws  = (float*)d_ws;

    float* s     = ws + OFF_S;
    float* kv    = ws + OFF_KV;
    float* q     = ws + OFF_Q;
    float* g1b   = ws + OFF_G1;
    float* g2b   = ws + OFF_G2;
    float* opre  = ws + OFF_OPRE;
    float* lr    = ws + OFF_LR;
    float* gate  = ws + OFF_GATE;
    float* sch   = ws + OFF_SCH;
    float* momg  = ws + OFF_MOMG;
    float* omd   = ws + OFF_OMD;

    // 1) store/retrieve norm (shared): s = rmsnorm(seq)
    k_rmsnorm<<<32768, 128, 0, stream>>>(seq, s);
    // 2) big projections via WMMA GEMM
    k_gemm<<<dim3(16, 512), 256, 0, stream>>>(s, Wkv, kv, 1024, 512, 0);  // kv
    k_gemm<<<dim3(8, 512), 256, 0, stream>>>(s, Wq, q, 512, 512, 0);      // q
    // 3) small per-head projections
    k_lr_gate<<<1024, 256, 0, stream>>>(s, Wstep, Wgate, lr, gate);
    k_chunk_mean<<<1024, 256, 0, stream>>>(s, sch);
    k_gates<<<16, 256, 0, stream>>>(sch, Wmom, Wdec, momg, omd);
    // 4) per-chunk MLP gradients (WMMA + TDM)
    k_chunk_grad<<<4096, 256, 0, stream>>>(kv, lr, w1, w2, g1b, g2b);
    // 5) momentum + decay scans over chunks (in place, prefetched)
    k_scan<<<512, 256, 0, stream>>>(g1b, momg, omd);
    k_scan<<<512, 256, 0, stream>>>(g2b, momg, omd);
    // 6) retrieve with fast weights (WMMA + TDM)
    k_retrieve<<<4096, 256, 0, stream>>>(q, w1, w2, g1b, g2b, gate, gamma, opre);
    // 7) combine heads + causal shift fused into GEMM store
    k_zero_head<<<504, 256, 0, stream>>>(out);
    k_gemm<<<dim3(8, 512), 256, 0, stream>>>(opre, Wcomb, out, 512, 512, 1);
}
